// InferenceDecoderCell_3770981286143
// MI455X (gfx1250) — matrix operand order repaired and checked by reading, not compiled, not run on hardware
//
#include <hip/hip_runtime.h>
#include <hip/hip_bf16.h>
#include <math.h>

typedef __attribute__((ext_vector_type(16))) __bf16          v16bf;
typedef __attribute__((ext_vector_type(16))) unsigned short  v16us;
typedef __attribute__((ext_vector_type(8)))  unsigned short  v8us;
typedef __attribute__((ext_vector_type(8)))  float           v8f;
typedef __attribute__((ext_vector_type(4)))  float           v4f;

__device__ __forceinline__ unsigned short f32_bf16_rne(float f) {
  union { float f; unsigned int u; } x; x.f = f;
  unsigned int u = x.u;
  return (unsigned short)((u + 0x7FFFu + ((u >> 16) & 1u)) >> 16);
}

// Pack two fp32 -> two bf16 in one dword (round-half-up + v_perm_b32).
__device__ __forceinline__ unsigned int pack2_bf16(float f0, float f1) {
  union { float f; unsigned int u; } a, b; a.f = f0; b.f = f1;
  unsigned int u0 = a.u + 0x8000u;
  unsigned int u1 = b.u + 0x8000u;
  // result bytes: [0]=u0.b2 [1]=u0.b3 [2]=u1.b2 [3]=u1.b3
  return __builtin_amdgcn_perm(u1, u0, 0x07060302u);
}

// ---------------------------------------------------------------------------
// GEMM: C[M,N] = epilogue( A[M,K] * B[K,N] )
//   epilogue: val = alpha*(acc + bias[n]); if(relu) max(0); if(res) += res[m,n]
// Block = 256 threads (8 waves) -> 128x32 C tile: wave w owns rows w*16..w*16+15
// and both 16-col halves (two accumulators sharing one A fragment).
// Per K-step(32): B 32x32 tile staged to LDS as bf16 (shared by all 8 waves),
// A fragment packed in-regs via v_perm, two v_wmma_f32_16x16x32_bf16 per wave.
// If seq_len>0 rows are (b*seq_len+s); tiles with s_start > *step are skipped.
// ---------------------------------------------------------------------------
#define BT_PITCH 40   // ushorts per LDS row -> 80B rows, 16B-aligned chunks
__global__ __launch_bounds__(256) void gemm_bf16_wmma(
    const float* __restrict__ A, const float* __restrict__ Bm,
    const float* __restrict__ bias, const float* __restrict__ res,
    float* __restrict__ C, int M, int N, int K,
    float alpha, int relu, int seq_len, const int* __restrict__ step_ptr)
{
  __shared__ unsigned short bT[32 * BT_PITCH];   // [n(0..31)][k(0..31)] n-major
  const int tid  = threadIdx.x;
  const int lane = tid & 31;
  const int wid  = tid >> 5;
  const int Mt = M >> 4;
  const int MB = (Mt + 7) >> 3;                  // 128-row block groups
  const int bm = blockIdx.x % MB;
  const int bn = blockIdx.x / MB;
  const int m0b = bm << 7;                       // block row base
  const int n0  = bn << 5;                       // block col base (32 cols)

  if (seq_len > 0) {                             // uniform early-out (pre-barrier)
    if ((m0b % seq_len) > *step_ptr) return;
  }

  const int mt = (m0b >> 4) + wid;               // this wave's 16-row tile
  bool wvalid = (mt < Mt);
  if (wvalid && seq_len > 0)
    wvalid = (((mt << 4) % seq_len) <= *step_ptr);

  const int half = lane >> 4;                    // 0: lanes 0-15, 1: lanes 16-31
  const int l15  = lane & 15;
  const int arow = (mt << 4) + l15;              // A fragment row
  const int nc0  = n0 + l15;                     // C columns (two halves)
  const int nc1  = n0 + 16 + l15;

  v8f c0 = {0.f, 0.f, 0.f, 0.f, 0.f, 0.f, 0.f, 0.f};
  v8f c1 = {0.f, 0.f, 0.f, 0.f, 0.f, 0.f, 0.f, 0.f};

  for (int k0 = 0; k0 < K; k0 += 32) {
    __syncthreads();
    #pragma unroll
    for (int j = 0; j < 4; ++j) {                // stage 32x32 B tile, coalesced
      const int idx = tid + j * 256;
      const int r  = idx >> 5;                   // k row   0..31
      const int cc = idx & 31;                   // n col   0..31
      bT[cc * BT_PITCH + r] = f32_bf16_rne(Bm[(size_t)(k0 + r) * N + n0 + cc]);
    }
    __syncthreads();

    if (!wvalid) continue;

    // A 16x32 bf16 fragment: elem i(0..7)->k=k0+8h+i ; i(8..15)->k=k0+16+8h+(i-8)
    const float* ap = A + (size_t)arow * K + k0 + 8 * half;
    __builtin_prefetch(ap + 32, 0, 0);
    const float4 a0 = *(const float4*)(ap);
    const float4 a1 = *(const float4*)(ap + 4);
    const float4 a2 = *(const float4*)(ap + 16);
    const float4 a3 = *(const float4*)(ap + 20);
    union { v16us us; unsigned int d[8]; } af;
    af.d[0] = pack2_bf16(a0.x, a0.y);
    af.d[1] = pack2_bf16(a0.z, a0.w);
    af.d[2] = pack2_bf16(a1.x, a1.y);
    af.d[3] = pack2_bf16(a1.z, a1.w);
    af.d[4] = pack2_bf16(a2.x, a2.y);
    af.d[5] = pack2_bf16(a2.z, a2.w);
    af.d[6] = pack2_bf16(a3.x, a3.y);
    af.d[7] = pack2_bf16(a3.z, a3.w);
    v16bf a = __builtin_bit_cast(v16bf, af.us);

    // B 32x16 fragments from LDS: lane n = l15 (+16), i(0..7)->k=8*half+i ; i(8..15)->k=16+8*half+(i-8)
    union { v16us us; v8us h[2]; } bf0, bf1;
    bf0.h[0] = *(const v8us*)&bT[l15 * BT_PITCH + 8 * half];
    bf0.h[1] = *(const v8us*)&bT[l15 * BT_PITCH + 8 * half + 16];
    bf1.h[0] = *(const v8us*)&bT[(16 + l15) * BT_PITCH + 8 * half];
    bf1.h[1] = *(const v8us*)&bT[(16 + l15) * BT_PITCH + 8 * half + 16];

    c0 = __builtin_amdgcn_wmma_f32_16x16x32_bf16(
        false, a, false, __builtin_bit_cast(v16bf, bf0.us), (short)0, c0, false, false);
    c1 = __builtin_amdgcn_wmma_f32_16x16x32_bf16(
        false, a, false, __builtin_bit_cast(v16bf, bf1.us), (short)0, c1, false, false);
  }

  if (!wvalid) return;
  const float bv0 = bias ? bias[nc0] : 0.0f;
  const float bv1 = bias ? bias[nc1] : 0.0f;
  const int m0 = mt << 4;
  #pragma unroll
  for (int j = 0; j < 8; ++j) {                  // C: VGPR j -> m = m0 + j + 8*half
    const int m = m0 + j + 8 * half;
    float val0 = (c0[j] + bv0) * alpha;
    float val1 = (c1[j] + bv1) * alpha;
    if (relu) { val0 = fmaxf(val0, 0.0f); val1 = fmaxf(val1, 0.0f); }
    if (res)  { val0 += res[(size_t)m * N + nc0]; val1 += res[(size_t)m * N + nc1]; }
    C[(size_t)m * N + nc0] = val0;
    C[(size_t)m * N + nc1] = val1;
  }
}

// ---------------------------------------------------------------------------
// Single-query attention per (b,h): scores -> softmax -> ctx = att @ V
// q: (B, 512) (already scaled by 1/sqrt(kd)); kbuf/vbuf: (B*S, 512); ctx: (B,512)
// ---------------------------------------------------------------------------
__global__ __launch_bounds__(256) void attn_kernel(
    const float* __restrict__ q, const float* __restrict__ kbuf,
    const float* __restrict__ vbuf, float* __restrict__ ctx,
    int S, const int* __restrict__ step_ptr)
{
  __shared__ float sc[256];
  __shared__ float red[256];
  __shared__ float qs[64];
  const int tid = threadIdx.x;
  const int b = blockIdx.x >> 3;
  const int h = blockIdx.x & 7;
  const int s_valid = step_ptr ? (*step_ptr + 1) : S;

  if (tid < 64) qs[tid] = q[b * 512 + h * 64 + tid];
  __syncthreads();

  for (int s = tid; s < S; s += 256) {
    float d = -1e30f;
    if (s < s_valid) {
      const float* kp = kbuf + (size_t)(b * S + s) * 512 + h * 64;
      float acc = 0.f;
      #pragma unroll
      for (int i = 0; i < 64; ++i) acc += qs[i] * kp[i];
      d = acc;
    }
    sc[s] = d;
  }
  __syncthreads();

  float lm = -1e30f;
  for (int s = tid; s < S; s += 256) lm = fmaxf(lm, sc[s]);
  red[tid] = lm; __syncthreads();
  for (int st = 128; st > 0; st >>= 1) {
    if (tid < st) red[tid] = fmaxf(red[tid], red[tid + st]);
    __syncthreads();
  }
  const float mx = red[0];
  __syncthreads();

  float ls = 0.f;
  for (int s = tid; s < S; s += 256) {
    if (s < s_valid) { float e = __expf(sc[s] - mx); sc[s] = e; ls += e; }
  }
  __syncthreads();
  red[tid] = ls; __syncthreads();
  for (int st = 128; st > 0; st >>= 1) {
    if (tid < st) red[tid] += red[tid + st];
    __syncthreads();
  }
  const float inv = 1.0f / red[0];
  __syncthreads();

  // ctx: 4 s-partitions x 64 dims
  {
    const int d = tid & 63;
    const int g = tid >> 6;          // 0..3
    const float* vp = vbuf + (size_t)(b * S) * 512 + h * 64 + d;
    float acc = 0.f;
    for (int s = g; s < s_valid; s += 4) acc += sc[s] * vp[(size_t)s * 512];
    red[tid] = acc;
  }
  __syncthreads();
  if (tid < 64) {
    const float tot = red[tid] + red[tid + 64] + red[tid + 128] + red[tid + 192];
    ctx[b * 512 + h * 64 + tid] = tot * inv;
  }
}

// ---------------------------------------------------------------------------
// LayerNorm over D=512, eps=1e-3, population variance.
// ---------------------------------------------------------------------------
__global__ __launch_bounds__(256) void ln_kernel(
    const float* __restrict__ in, const float* __restrict__ g,
    const float* __restrict__ bb, float* __restrict__ out)
{
  __shared__ float s1[256], s2[256];
  const int tid = threadIdx.x, row = blockIdx.x;
  const float* ip = in + (size_t)row * 512;
  const float a = ip[tid], c = ip[tid + 256];
  s1[tid] = a + c; s2[tid] = a * a + c * c;
  __syncthreads();
  for (int st = 128; st > 0; st >>= 1) {
    if (tid < st) { s1[tid] += s1[tid + st]; s2[tid] += s2[tid + st]; }
    __syncthreads();
  }
  const float mean = s1[0] * (1.0f / 512.0f);
  const float var  = s2[0] * (1.0f / 512.0f) - mean * mean;
  const float rstd = rsqrtf(var + 0.001f);
  float* op = out + (size_t)row * 512;
  op[tid]       = (a - mean) * rstd * g[tid] + bb[tid];
  op[tid + 256] = (c - mean) * rstd * g[tid + 256] + bb[tid + 256];
}

// ---------------------------------------------------------------------------
// x[b,d] = emb[token[b], d] + PE(step, d)
// ---------------------------------------------------------------------------
__global__ void embed_kernel(const int* __restrict__ token,
                             const int* __restrict__ step,
                             const float* __restrict__ emb,
                             float* __restrict__ x)
{
  const int b = blockIdx.x, tid = threadIdx.x;
  const int t = token[b];
  const float st = (float)(*step);
  for (int d = tid; d < 512; d += 256) {
    const float expo = (float)(2 * (d / 2)) * (1.0f / 512.0f);
    const float ang = st * __expf(-expo * 9.21034037198f); // /10000^expo
    const float pe = (d & 1) ? __cosf(ang) : __sinf(ang);
    x[b * 512 + d] = emb[(size_t)t * 512 + d] + pe;
  }
}

// ---------------------------------------------------------------------------
// new_seqs[l] = cur_seqs[l]; new_seqs[l][:, step, :] = x
// Streaming copy: non-temporal on both sides to keep weights resident in the
// 192MB L2 (this kernel moves ~768MB/call and must not thrash the cache).
// ---------------------------------------------------------------------------
__global__ void store_seq_kernel(const float* __restrict__ cur_l,
                                 const float* __restrict__ x,
                                 const int* __restrict__ step,
                                 float* __restrict__ new_l)
{
  const size_t idx4 = (size_t)blockIdx.x * blockDim.x + threadIdx.x;
  const size_t flat = idx4 * 4;                 // B*S*D = 128*256*512 floats
  if (flat >= (size_t)128 * 256 * 512) return;
  const int d = (int)(flat & 511);
  const int s = (int)((flat >> 9) & 255);
  const int b = (int)(flat >> 17);
  v4f v;
  if (s == *step) v = *(const v4f*)(x + (size_t)b * 512 + d);
  else            v = __builtin_nontemporal_load((const v4f*)(cur_l + flat));
  __builtin_nontemporal_store(v, (v4f*)(new_l + flat));
}

// ---------------------------------------------------------------------------
extern "C" void kernel_launch(void* const* d_in, const int* in_sizes, int n_in,
                              void* d_out, int out_size, void* d_ws, size_t ws_size,
                              hipStream_t stream) {
  (void)in_sizes; (void)n_in; (void)out_size; (void)ws_size;
  constexpr int L = 6, B = 128, S = 256, D = 512, H = 8, F = 2048, V = 32000;

  const int*   token = (const int*)  d_in[0];
  const int*   step  = (const int*)  d_in[1];
  const float* enc   = (const float*)d_in[2];
  const float* cur   = (const float*)d_in[3];
  const float* emb   = (const float*)d_in[4];
  const float* Wq_s  = (const float*)d_in[5];
  const float* bq_s  = (const float*)d_in[6];
  const float* Wk_s  = (const float*)d_in[7];
  const float* bk_s  = (const float*)d_in[8];
  const float* Wv_s  = (const float*)d_in[9];
  const float* bv_s  = (const float*)d_in[10];
  const float* Wo_s  = (const float*)d_in[11];
  const float* bo_s  = (const float*)d_in[12];
  const float* ln1_g = (const float*)d_in[13];
  const float* ln1_b = (const float*)d_in[14];
  const float* Wq_c  = (const float*)d_in[15];
  const float* bq_c  = (const float*)d_in[16];
  const float* Wk_c  = (const float*)d_in[17];
  const float* bk_c  = (const float*)d_in[18];
  const float* Wv_c  = (const float*)d_in[19];
  const float* bv_c  = (const float*)d_in[20];
  const float* Wo_c  = (const float*)d_in[21];
  const float* bo_c  = (const float*)d_in[22];
  const float* ln2_g = (const float*)d_in[23];
  const float* ln2_b = (const float*)d_in[24];
  const float* W1    = (const float*)d_in[25];
  const float* b1    = (const float*)d_in[26];
  const float* W2    = (const float*)d_in[27];
  const float* b2    = (const float*)d_in[28];
  const float* ln3_g = (const float*)d_in[29];
  const float* ln3_b = (const float*)d_in[30];
  const float* Wlog  = (const float*)d_in[31];
  const float* blog  = (const float*)d_in[32];

  float* out_logit = (float*)d_out;                     // (B, V)
  float* new_seqs  = out_logit + (size_t)B * V;         // (L, B, S, D)

  float* ws = (float*)d_ws;
  size_t o = 0;
  float* kbuf = ws + o; o += (size_t)B * S * D;   // 64 MB
  float* vbuf = ws + o; o += (size_t)B * S * D;   // 64 MB
  float* qb   = ws + o; o += (size_t)B * D;
  float* ctx  = ws + o; o += (size_t)B * D;
  float* x    = ws + o; o += (size_t)B * D;
  float* x1   = ws + o; o += (size_t)B * D;
  float* x2   = ws + o; o += (size_t)B * D;
  float* tmp  = ws + o; o += (size_t)B * D;
  float* hbuf = ws + o; o += (size_t)B * F;

  auto GEMM = [&](const float* A, const float* Bm, const float* bias,
                  const float* res, float* C, int M, int N, int K,
                  float alpha, int relu, int seq_len, const int* sp) {
    const int Mt = M / 16;
    const int MB = (Mt + 7) / 8;
    const int NB = N / 32;
    gemm_bf16_wmma<<<MB * NB, 256, 0, stream>>>(
        A, Bm, bias, res, C, M, N, K, alpha, relu, seq_len, sp);
  };

  embed_kernel<<<B, 256, 0, stream>>>(token, step, emb, x);

  const float qscale = 0.125f;   // 1/sqrt(64)
  for (int l = 0; l < L; ++l) {
    const float* cur_l = cur + (size_t)l * B * S * D;
    float* new_l = new_seqs + (size_t)l * B * S * D;
    store_seq_kernel<<<(B * S * D / 4 + 255) / 256, 256, 0, stream>>>(
        cur_l, x, step, new_l);

    // ---- self attention (keys = seq prefix, rows s<=step) ----
    GEMM(x,     Wq_s + (size_t)l * D * D, bq_s + l * D, nullptr, qb,   B,     D, D, qscale, 0, 0, nullptr);
    GEMM(new_l, Wk_s + (size_t)l * D * D, bk_s + l * D, nullptr, kbuf, B * S, D, D, 1.0f,   0, S, step);
    GEMM(new_l, Wv_s + (size_t)l * D * D, bv_s + l * D, nullptr, vbuf, B * S, D, D, 1.0f,   0, S, step);
    attn_kernel<<<B * H, 256, 0, stream>>>(qb, kbuf, vbuf, ctx, S, step);
    GEMM(ctx,   Wo_s + (size_t)l * D * D, bo_s + l * D, x,       tmp,  B,     D, D, 1.0f,   0, 0, nullptr);
    ln_kernel<<<B, 256, 0, stream>>>(tmp, ln1_g + l * D, ln1_b + l * D, x1);

    // ---- cross attention (keys = enc_outputs, all 256) ----
    GEMM(x1,  Wq_c + (size_t)l * D * D, bq_c + l * D, nullptr, qb,   B,     D, D, qscale, 0, 0, nullptr);
    GEMM(enc, Wk_c + (size_t)l * D * D, bk_c + l * D, nullptr, kbuf, B * S, D, D, 1.0f,   0, 0, nullptr);
    GEMM(enc, Wv_c + (size_t)l * D * D, bv_c + l * D, nullptr, vbuf, B * S, D, D, 1.0f,   0, 0, nullptr);
    attn_kernel<<<B * H, 256, 0, stream>>>(qb, kbuf, vbuf, ctx, S, nullptr);
    GEMM(ctx, Wo_c + (size_t)l * D * D, bo_c + l * D, x1,      tmp,  B,     D, D, 1.0f,   0, 0, nullptr);
    ln_kernel<<<B, 256, 0, stream>>>(tmp, ln2_g + l * D, ln2_b + l * D, x2);

    // ---- FFN ----
    GEMM(x2,   W1 + (size_t)l * D * F, b1 + l * F, nullptr, hbuf, B, F, D, 1.0f, 1, 0, nullptr);
    GEMM(hbuf, W2 + (size_t)l * F * D, b2 + l * D, x2,      tmp,  B, D, F, 1.0f, 0, 0, nullptr);
    ln_kernel<<<B, 256, 0, stream>>>(tmp, ln3_g + l * D, ln3_b + l * D, x);
  }

  // ---- logits ----
  GEMM(x, Wlog, blog, nullptr, out_logit, B, V, D, 1.0f, 0, 0, nullptr);
}
